// WindowAttention_8744553414870
// MI455X (gfx1250) — compile-verified, matrix operand order checked by reading, not run on hardware
//
#include <hip/hip_runtime.h>
#include <hip/hip_bf16.h>

#define DIM   128
#define HEADS 4
#define HD    32
#define NTOK  64
#define NWIN  4096

typedef _Float16 h16;
typedef __attribute__((ext_vector_type(16))) _Float16 v16h;
typedef __attribute__((ext_vector_type(8)))  _Float16 v8h;
typedef __attribute__((ext_vector_type(8)))  float    v8f;
typedef __attribute__((ext_vector_type(4)))  float    v4f;

// LDS strides in halfs (all multiples of 8 -> 16B aligned rows)
#define XW_S 136   // x window: 64 x 136
#define QK_S 264   // Q|K:      64 x 264 (cols 0..127 = Q*scale, 128..255 = K)
#define VT_S 72    // V^T per head: 32 x 72  ([d][token])
#define PH_S 72    // P per head:   64 x 72
#define OB_S 136   // O:        64 x 136

// Load a 16x32 f16 A/B fragment row-slice: 8 contiguous halfs at p and p+16.
// Caller computes p = base + row*stride + kbase  (row = lane&15 dependent,
// kbase = k0 + 8*(lane>>4)) per the CDNA5 16-bit WMMA operand layout.
__device__ inline v16h frag16(const h16* __restrict__ p) {
    v8h lo = *(const v8h*)p;
    v8h hi = *(const v8h*)(p + 16);
    v16h r;
#pragma unroll
    for (int i = 0; i < 8; ++i) { r[i] = lo[i]; r[8 + i] = hi[i]; }
    return r;
}

__device__ inline v8f wmma_f16(v16h a, v16h b, v8f c) {
    return __builtin_amdgcn_wmma_f32_16x16x32_f16(false, a, false, b,
                                                  (short)0, c, false, false);
}

__global__ __launch_bounds__(256)
void swin_prep_weights(const float* __restrict__ wqkv,
                       const float* __restrict__ wproj,
                       h16* __restrict__ wqkv_h,
                       h16* __restrict__ wproj_h) {
    int i = blockIdx.x * 256 + threadIdx.x;
    if (i < 3 * DIM * DIM) wqkv_h[i] = (h16)wqkv[i];
    if (i < DIM * DIM)     wproj_h[i] = (h16)wproj[i];
}

__global__ __launch_bounds__(256)
void swin_window_attn(const float* __restrict__ x,
                      const float* __restrict__ mask,
                      const float* __restrict__ b_qkv,
                      const float* __restrict__ b_proj,
                      const h16* __restrict__ wqkv_h,   // [384][128] f16
                      const h16* __restrict__ wproj_h,  // [128][128] f16
                      float* __restrict__ out,          // [8192][64][128]
                      float* __restrict__ attn_out) {   // [8192][4][64][64]
    // LDS with phase-based reuse:
    //   region A [0,36864):      Xw (ph1-2, 17408B)  then Ph (ph3-4, 36864B)
    //   region B [36864,70656):  QK (ph2-3, 33792B)  then Ob (ph4-5, 17408B)
    //   region C [70656,89088):  Vt (ph2-4, 18432B)
    __shared__ __align__(16) char smem[89088];
    h16* Xw = (h16*)smem;
    h16* Ph = (h16*)smem;
    h16* QK = (h16*)(smem + 36864);
    h16* Ob = (h16*)(smem + 36864);
    h16* Vt = (h16*)(smem + 70656);

    const int b    = blockIdx.x;
    const int tid  = threadIdx.x;
    const int lane = tid & 31;
    const int wv   = tid >> 5;      // 0..7
    const int lm   = lane & 15;
    const int lhi  = lane >> 4;     // 0/1

    // ---------- phase 1: stage x window as f16 (streaming, read-once) ----------
    const float* xsrc = x + (size_t)b * NTOK * DIM;
    for (int i = tid * 4; i < NTOK * DIM; i += 256 * 4) {
        const v4f v4 = __builtin_nontemporal_load((const v4f*)(xsrc + i));
        int m = i >> 7, k = i & (DIM - 1);
        Xw[m * XW_S + k + 0] = (h16)v4.x;
        Xw[m * XW_S + k + 1] = (h16)v4.y;
        Xw[m * XW_S + k + 2] = (h16)v4.z;
        Xw[m * XW_S + k + 3] = (h16)v4.w;
    }
    __syncthreads();

    const float scale = 0.17677669529663687f;  // 1/sqrt(32)

    // ---------- phase 2: QKV = Xw(64x128) @ Wqkv^T(128x384) + b ----------
    // wave wv: n-tiles {3wv..3wv+2}, all 4 m-tiles, K = 4 steps of 32
    for (int nti = 0; nti < 3; ++nti) {
        const int n0 = (3 * wv + nti) * 16;
        v8f acc[4] = {};
#pragma unroll
        for (int ks = 0; ks < 4; ++ks) {
            const int kb = ks * 32 + (lhi << 3);
            v16h bf = frag16(wqkv_h + (size_t)(n0 + lm) * DIM + kb);
#pragma unroll
            for (int mt = 0; mt < 4; ++mt) {
                v16h af = frag16(Xw + (mt * 16 + lm) * XW_S + kb);
                acc[mt] = wmma_f16(af, bf, acc[mt]);
            }
        }
        const float bias = b_qkv[n0 + lm];
        const int n = n0 + lm;
#pragma unroll
        for (int mt = 0; mt < 4; ++mt) {
#pragma unroll
            for (int r = 0; r < 8; ++r) {
                const int m = mt * 16 + r + (lhi << 3);
                const float v = acc[mt][r] + bias;
                if (n < 128) {                 // Q (pre-scaled)
                    QK[m * QK_S + n] = (h16)(v * scale);
                } else if (n < 256) {          // K
                    QK[m * QK_S + n] = (h16)v;
                } else {                       // V, transposed per head
                    const int c = n - 256;
                    Vt[(c >> 5) * (HD * VT_S) + (c & 31) * VT_S + m] = (h16)v;
                }
            }
        }
    }
    __syncthreads();

    // ---------- phase 3: S = Qh @ Kh^T, +mask, softmax ----------
    const int h   = wv >> 1;          // head handled by this wave pair
    const int sub = wv & 1;           // row half: rows 32*sub .. 32*sub+31
    const int qc  = h * HD;           // Q col base in QK
    const int kc  = 128 + h * HD;     // K col base in QK

    v8f s[2][4] = {};
    {
        const int kb = lhi << 3;      // K = 32, single WMMA step
#pragma unroll
        for (int ntl = 0; ntl < 4; ++ntl) {
            v16h bf = frag16(QK + (ntl * 16 + lm) * QK_S + kc + kb);
#pragma unroll
            for (int mtl = 0; mtl < 2; ++mtl) {
                const int m0 = (sub * 2 + mtl) * 16;
                v16h af = frag16(QK + (m0 + lm) * QK_S + qc + kb);
                s[mtl][ntl] = wmma_f16(af, bf, s[mtl][ntl]);
            }
        }
    }
    // add mask (window id = b % NWIN); mask is re-read by b+4096 -> keep temporal
    const float* mptr = mask + (size_t)(b & (NWIN - 1)) * NTOK * NTOK;
#pragma unroll
    for (int mtl = 0; mtl < 2; ++mtl) {
#pragma unroll
        for (int r = 0; r < 8; ++r) {
            const int m = (sub * 2 + mtl) * 16 + r + (lhi << 3);
#pragma unroll
            for (int ntl = 0; ntl < 4; ++ntl)
                s[mtl][ntl][r] += mptr[m * NTOK + ntl * 16 + lm];
        }
    }
    // row softmax: a row lives in one VGPR slot across 16 lanes x 4 accums
#pragma unroll
    for (int mtl = 0; mtl < 2; ++mtl) {
#pragma unroll
        for (int r = 0; r < 8; ++r) {
            float mx = fmaxf(fmaxf(s[mtl][0][r], s[mtl][1][r]),
                             fmaxf(s[mtl][2][r], s[mtl][3][r]));
#pragma unroll
            for (int off = 1; off < 16; off <<= 1)
                mx = fmaxf(mx, __shfl_xor(mx, off, 32));
            float sum = 0.f;
#pragma unroll
            for (int ntl = 0; ntl < 4; ++ntl) {
                float e = __expf(s[mtl][ntl][r] - mx);
                s[mtl][ntl][r] = e;
                sum += e;
            }
#pragma unroll
            for (int off = 1; off < 16; off <<= 1)
                sum += __shfl_xor(sum, off, 32);
            const float inv = 1.0f / sum;
#pragma unroll
            for (int ntl = 0; ntl < 4; ++ntl)
                s[mtl][ntl][r] *= inv;
        }
    }
    // emit attn_map (f32, global, streamed NT) and P (f16 -> LDS over dead Xw)
    float* aout = attn_out + ((size_t)b * HEADS + h) * NTOK * NTOK;
#pragma unroll
    for (int mtl = 0; mtl < 2; ++mtl) {
#pragma unroll
        for (int r = 0; r < 8; ++r) {
            const int m = (sub * 2 + mtl) * 16 + r + (lhi << 3);
#pragma unroll
            for (int ntl = 0; ntl < 4; ++ntl) {
                const int n = ntl * 16 + lm;
                const float p = s[mtl][ntl][r];
                __builtin_nontemporal_store(p, aout + m * NTOK + n);
                Ph[h * (NTOK * PH_S) + m * PH_S + n] = (h16)p;
            }
        }
    }
    __syncthreads();   // all QK reads done -> Ob may overlay QK

    // ---------- phase 4: O = P(64x64) @ Vh(64x32) ----------
    v8f o[2][2] = {};
#pragma unroll
    for (int ks = 0; ks < 2; ++ks) {
        const int kb = ks * 32 + (lhi << 3);
#pragma unroll
        for (int ntl = 0; ntl < 2; ++ntl) {
            // B[k][n] = Vh[token k][d n] = Vt[n][k] (contiguous in k)
            v16h bf = frag16(Vt + h * (HD * VT_S) + (ntl * 16 + lm) * VT_S + kb);
#pragma unroll
            for (int mtl = 0; mtl < 2; ++mtl) {
                const int m0 = (sub * 2 + mtl) * 16;
                v16h af = frag16(Ph + h * (NTOK * PH_S) + (m0 + lm) * PH_S + kb);
                o[mtl][ntl] = wmma_f16(af, bf, o[mtl][ntl]);
            }
        }
    }
#pragma unroll
    for (int mtl = 0; mtl < 2; ++mtl) {
#pragma unroll
        for (int r = 0; r < 8; ++r) {
            const int m = (sub * 2 + mtl) * 16 + r + (lhi << 3);
#pragma unroll
            for (int ntl = 0; ntl < 2; ++ntl)
                Ob[m * OB_S + h * HD + ntl * 16 + lm] = (h16)o[mtl][ntl][r];
        }
    }
    __syncthreads();

    // ---------- phase 5: out = Ob(64x128) @ Wproj^T(128x128) + b ----------
    {
        const int n0 = wv * 16;               // 8 waves x 16 cols = 128
        v8f acc[4] = {};
#pragma unroll
        for (int ks = 0; ks < 4; ++ks) {
            const int kb = ks * 32 + (lhi << 3);
            v16h bf = frag16(wproj_h + (size_t)(n0 + lm) * DIM + kb);
#pragma unroll
            for (int mt = 0; mt < 4; ++mt) {
                v16h af = frag16(Ob + (mt * 16 + lm) * OB_S + kb);
                acc[mt] = wmma_f16(af, bf, acc[mt]);
            }
        }
        const float bias = b_proj[n0 + lm];
        float* optr = out + (size_t)b * NTOK * DIM;
#pragma unroll
        for (int mt = 0; mt < 4; ++mt) {
#pragma unroll
            for (int r = 0; r < 8; ++r) {
                const int m = mt * 16 + r + (lhi << 3);
                __builtin_nontemporal_store(acc[mt][r] + bias,
                                            optr + m * DIM + n0 + lm);
            }
        }
    }
}

extern "C" void kernel_launch(void* const* d_in, const int* in_sizes, int n_in,
                              void* d_out, int out_size, void* d_ws, size_t ws_size,
                              hipStream_t stream) {
    const float* x      = (const float*)d_in[0];
    const float* mask   = (const float*)d_in[1];
    const float* w_qkv  = (const float*)d_in[2];
    const float* b_qkv  = (const float*)d_in[3];
    const float* w_proj = (const float*)d_in[4];
    const float* b_proj = (const float*)d_in[5];

    h16* wqkv_h  = (h16*)d_ws;                          // 384*128 halfs = 98304 B
    h16* wproj_h = (h16*)((char*)d_ws + 3 * DIM * DIM * sizeof(h16));

    float* out_ptr  = (float*)d_out;                    // 8192*64*128 floats
    float* attn_ptr = out_ptr + (size_t)8192 * NTOK * DIM;

    swin_prep_weights<<<(3 * DIM * DIM + 255) / 256, 256, 0, stream>>>(
        w_qkv, w_proj, wqkv_h, wproj_h);

    swin_window_attn<<<8192, 256, 0, stream>>>(
        x, mask, b_qkv, b_proj, wqkv_h, wproj_h, out_ptr, attn_ptr);
}
